// GAEEncoder_39367670235137
// MI455X (gfx1250) — matrix operand order repaired and checked by reading, not compiled, not run on hardware
//
#include <hip/hip_runtime.h>
#include <hip/hip_bf16.h>

typedef __attribute__((ext_vector_type(16))) _Float16 v16h;
typedef __attribute__((ext_vector_type(8)))  float    v8f;

union h2u { unsigned u; _Float16 h[2]; };

// ---------------------------------------------------------------------------
// Utility: grid-stride zero
// ---------------------------------------------------------------------------
__global__ void zero_f32(float* __restrict__ p, long long n) {
    long long i = (long long)blockIdx.x * blockDim.x + threadIdx.x;
    long long stride = (long long)gridDim.x * blockDim.x;
    for (; i < n; i += stride) p[i] = 0.0f;
}

// ---------------------------------------------------------------------------
// Degree: count edges at destination (self-loop added in transform)
// ---------------------------------------------------------------------------
__global__ void deg_count(const int* __restrict__ dst, float* __restrict__ deg, int E) {
    int e = blockIdx.x * blockDim.x + threadIdx.x;
    if (e < E) atomicAdd(&deg[dst[e]], 1.0f);
}

__global__ void deg_to_dis(float* __restrict__ deg, int N) {
    int i = blockIdx.x * blockDim.x + threadIdx.x;
    if (i < N) deg[i] = rsqrtf(deg[i] + 1.0f);  // +1 self loop
}

// ---------------------------------------------------------------------------
// Pre-pack weights: f32 [K][NC] -> f16 pairs along K: Wp[k/2][n] = {W[k][n], W[k+1][n]}
// One-time conversion so the GEMM inner loop loads B VGPRs as single b32s.
// ---------------------------------------------------------------------------
__global__ void pack_w_f16(const float* __restrict__ W, unsigned* __restrict__ Wp,
                           int K, int NC) {
    int i = blockIdx.x * blockDim.x + threadIdx.x;
    int total = (K / 2) * NC;
    if (i >= total) return;
    int kp = i / NC, n = i % NC;
    h2u t;
    t.h[0] = (_Float16)W[(size_t)(2 * kp) * NC + n];
    t.h[1] = (_Float16)W[(size_t)(2 * kp + 1) * NC + n];
    Wp[i] = t.u;
}

// ---------------------------------------------------------------------------
// Precompute symmetric edge norm: norm[e] = dis[src]*dis[dst] (shared by layers)
// ---------------------------------------------------------------------------
__global__ void edge_norm(const int* __restrict__ src, const int* __restrict__ dst,
                          const float* __restrict__ dis, float* __restrict__ norm, int E) {
    int e = blockIdx.x * blockDim.x + threadIdx.x;
    if (e < E) norm[e] = dis[src[e]] * dis[dst[e]];
}

// ---------------------------------------------------------------------------
// WMMA GEMM: H[M,NC] = X[M,K] * W[K,NC]  (f32 X staged to LDS as f16,
// pre-packed f16 W, f32 accumulate). One wave per 16x16 output tile;
// NC/16 waves per block cover a 16-row stripe x all columns.
// ---------------------------------------------------------------------------
template <int K, int NC>
__global__ __launch_bounds__(32 * (NC / 16))
void gemm_wmma(const float* __restrict__ X, const unsigned* __restrict__ Wp,
               float* __restrict__ H, int M) {
    __shared__ _Float16 As[16 * K];  // 16 rows x K, row-major, f16

    const int row0 = blockIdx.x * 16;
    const int tid  = threadIdx.x;
    const int nthr = 32 * (NC / 16);

    // Stage A tile: coalesced float4 global loads, convert to f16 in LDS.
    const int total4 = 16 * K / 4;
    for (int i = tid; i < total4; i += nthr) {
        const int flat = i * 4;
        const int r    = flat / K;
        const int kk   = flat % K;
        const int row  = row0 + r;
        float4 v = make_float4(0.f, 0.f, 0.f, 0.f);
        if (row < M) v = *(const float4*)(X + (size_t)row * K + kk);
        As[flat + 0] = (_Float16)v.x;
        As[flat + 1] = (_Float16)v.y;
        As[flat + 2] = (_Float16)v.z;
        As[flat + 3] = (_Float16)v.w;
    }
    __syncthreads();

    const int wave = tid >> 5;
    const int lane = tid & 31;
    const int g    = lane >> 4;   // lane half
    const int m    = lane & 15;   // A row / B-D column within tile
    const int col  = wave * 16 + m;

    v8f acc = {};
#pragma unroll
    for (int k0 = 0; k0 < K; k0 += 32) {
        v16h a, b;
        // A fragment (ISA 7.12.2, 16-bit A 16x32): lane half offsets K by 8
        // within each 16-K half; pairs are K-contiguous -> b32/b128 LDS loads.
#pragma unroll
        for (int j = 0; j < 8; ++j) {
            const int kb = k0 + 8 * g + 2 * j + ((j >= 4) ? 8 : 0);
            a[2 * j]     = As[m * K + kb];
            a[2 * j + 1] = As[m * K + kb + 1];
        }
        // B fragment (32x16): VGPR j holds K = 8*g + 2j (+8 for j>=4), +1; lane = column.
        // Pre-packed: one b32 load per VGPR, no conversion, lanes coalesced.
#pragma unroll
        for (int j = 0; j < 8; ++j) {
            const int kp = (k0 >> 1) + 4 * g + j + ((j >= 4) ? 4 : 0);  // (k0 + 8g + 2j + (j>=4?8:0)) / 2
            h2u t;
            t.u = Wp[(size_t)kp * NC + col];
            b[2 * j]     = t.h[0];
            b[2 * j + 1] = t.h[1];
        }
        acc = __builtin_amdgcn_wmma_f32_16x16x32_f16(
            /*neg_a=*/false, a, /*neg_b=*/false, b,
            /*c_mod=*/(short)0, acc, /*reuse_a=*/false, /*reuse_b=*/false);
    }

    // C/D layout: VGPR r -> row r + 8*g, column = lane&15.
#pragma unroll
    for (int r = 0; r < 8; ++r) {
        const int row = row0 + r + 8 * g;
        if (row < M) H[(size_t)row * NC + col] = acc[r];
    }
}

// ---------------------------------------------------------------------------
// Edge scatter: agg[dst] += h[src] * norm[e], 4 channels per thread.
// Consecutive threads cover consecutive channel chunks of one edge -> the
// h[src] row read is coalesced; agg updates go through L2 atomics.
// ---------------------------------------------------------------------------
template <int NC>
__global__ void scatter_add(const int* __restrict__ src, const int* __restrict__ dst,
                            const float* __restrict__ norm, const float* __restrict__ h,
                            float* __restrict__ agg, int E) {
    const long long tid = (long long)blockIdx.x * blockDim.x + threadIdx.x;
    const int C4 = NC / 4;
    const long long total = (long long)E * C4;
    if (tid >= total) return;
    const int e = (int)(tid / C4);
    const int c = (int)(tid % C4) * 4;
    const int s = src[e];
    const int d = dst[e];
    const float nm = norm[e];
    const float4 hv = *(const float4*)(h + (size_t)s * NC + c);
    float* ap = agg + (size_t)d * NC + c;
    atomicAdd(ap + 0, hv.x * nm);
    atomicAdd(ap + 1, hv.y * nm);
    atomicAdd(ap + 2, hv.z * nm);
    atomicAdd(ap + 3, hv.w * nm);
}

// ---------------------------------------------------------------------------
// Finalize: out = relu(agg + h*dis^2 + bias). May alias agg (elementwise).
// ---------------------------------------------------------------------------
template <int NC>
__global__ void finalize_relu(const float* __restrict__ h, const float* __restrict__ dis,
                              const float* __restrict__ bias, const float* __restrict__ agg,
                              float* __restrict__ out, int N) {
    const long long tid = (long long)blockIdx.x * blockDim.x + threadIdx.x;
    const long long total = (long long)N * NC;
    if (tid >= total) return;
    const int row = (int)(tid / NC);
    const int c   = (int)(tid % NC);
    const float d = dis[row];
    const float v = agg[tid] + h[tid] * d * d + bias[c];
    out[tid] = fmaxf(v, 0.0f);
}

// ---------------------------------------------------------------------------
// Launch
// ---------------------------------------------------------------------------
extern "C" void kernel_launch(void* const* d_in, const int* in_sizes, int n_in,
                              void* d_out, int out_size, void* d_ws, size_t ws_size,
                              hipStream_t stream) {
    const float* x  = (const float*)d_in[0];
    const int*   ei = (const int*)d_in[1];
    const float* W1 = (const float*)d_in[2];
    const float* b1 = (const float*)d_in[3];
    const float* W2 = (const float*)d_in[4];
    const float* b2 = (const float*)d_in[5];

    const int N = in_sizes[0] / 256;   // 50000
    const int E = in_sizes[1] / 2;     // 800000
    const int* src = ei;
    const int* dst = ei + E;

    // Workspace layout (floats/uints). Whole working set ~81 MB -> L2 resident.
    float* ws = (float*)d_ws;
    const size_t NAL = ((size_t)N + 63) & ~(size_t)63;
    float*    dis  = ws;
    float*    h1   = ws + NAL;                  // [N,128]
    float*    agg1 = h1 + (size_t)N * 128;      // [N,128]  (becomes y1 after finalize)
    float*    h2   = agg1 + (size_t)N * 128;    // [N,64]
    float*    agg2 = h2 + (size_t)N * 64;       // [N,64]
    float*    nrm  = agg2 + (size_t)N * 64;     // [E]
    unsigned* wpk1 = (unsigned*)(nrm + E);      // [128][128] packed f16 pairs
    unsigned* wpk2 = wpk1 + (256 / 2) * 128;    // [64][64]

    const int T = 256;
    // Zero accumulators (fresh every call: deterministic).
    zero_f32<<<1024, T, 0, stream>>>(dis, N);
    zero_f32<<<4096, T, 0, stream>>>(agg1, (long long)N * 128);
    zero_f32<<<4096, T, 0, stream>>>(agg2, (long long)N * 64);

    // One-time weight packing (f32 -> packed f16 K-pairs).
    pack_w_f16<<<((256 / 2) * 128 + T - 1) / T, T, 0, stream>>>(W1, wpk1, 256, 128);
    pack_w_f16<<<((128 / 2) * 64 + T - 1) / T, T, 0, stream>>>(W2, wpk2, 128, 64);

    // Degree -> D^{-1/2}, then per-edge norm (shared by both layers).
    deg_count<<<(E + T - 1) / T, T, 0, stream>>>(dst, dis, E);
    deg_to_dis<<<(N + T - 1) / T, T, 0, stream>>>(dis, N);
    edge_norm<<<(E + T - 1) / T, T, 0, stream>>>(src, dst, dis, nrm, E);

    // Layer 1: h1 = x @ W1 (WMMA), scatter, finalize (y1 written into agg1)
    gemm_wmma<256, 128><<<(N + 15) / 16, 256, 0, stream>>>(x, wpk1, h1, N);
    {
        const long long tot = (long long)E * (128 / 4);
        scatter_add<128><<<(unsigned)((tot + T - 1) / T), T, 0, stream>>>(src, dst, nrm, h1, agg1, E);
    }
    {
        const long long tot = (long long)N * 128;
        finalize_relu<128><<<(unsigned)((tot + T - 1) / T), T, 0, stream>>>(h1, dis, b1, agg1, agg1, N);
    }

    // Layer 2: h2 = y1 @ W2 (WMMA), scatter, finalize -> d_out
    gemm_wmma<128, 64><<<(N + 15) / 16, 128, 0, stream>>>(agg1, wpk2, h2, N);
    {
        const long long tot = (long long)E * (64 / 4);
        scatter_add<64><<<(unsigned)((tot + T - 1) / T), T, 0, stream>>>(src, dst, nrm, h2, agg2, E);
    }
    {
        const long long tot = (long long)N * 64;
        finalize_relu<64><<<(unsigned)((tot + T - 1) / T), T, 0, stream>>>(h2, dis, b2, agg2, (float*)d_out, N);
    }
}
